// DiffPool_61967788147142
// MI455X (gfx1250) — compile-verified, matrix operand order checked by reading, not run on hardware
//
#include <hip/hip_runtime.h>
#include <math.h>

// ---------------------------------------------------------------------------
// DiffPool forward for MI455X (gfx1250, wave32).
// All O(N^2) contractions run on v_wmma_f32_16x16x32_bf16. Operands are
// pre-converted to bf16 in K-contiguous ("TN") layout with K padded to a
// multiple of 32, so each lane builds its WMMA fragment from aligned b128
// loads. The GEMM computes a 32x32 output block per wave (2x2 WMMA tiles:
// each fragment feeds two WMMAs) and software-pipelines the K loop (next
// chunk's loads issue before the current chunk's WMMAs) so loads overlap
// the matrix pipe. Working set (~90MB) is L2-resident (192MB).
// ---------------------------------------------------------------------------

typedef __attribute__((ext_vector_type(16))) __bf16        v16bf;
typedef __attribute__((ext_vector_type(8)))  float         v8f;
typedef __attribute__((ext_vector_type(8)))  unsigned int  v8u;

static __device__ __forceinline__ float waveSum(float v) {
#pragma unroll
  for (int o = 16; o > 0; o >>= 1) v += __shfl_xor(v, o, 32);
  return v;
}
static __device__ __forceinline__ float waveMax(float v) {
#pragma unroll
  for (int o = 16; o > 0; o >>= 1) v = fmaxf(v, __shfl_xor(v, o, 32));
  return v;
}

// K offset of fragment dword r (0..7) for this lane-half, per ISA 7.12.2
// "16-bit A-Matrix 16x32": VGPR0:K=0,1|8,9 ... VGPR4:K=16,17|24,25 ...
static __device__ __forceinline__ int frag_k(int r, int half) {
  int kb = (r < 4) ? (2 * r) : (2 * r + 8);
  return kb + 8 * half;
}

// 8 aligned dword loads -> one 16x32 bf16 fragment for this lane.
static __device__ __forceinline__ v8u load_frag(const __bf16* base, int k0, int half) {
  v8u u;
#pragma unroll
  for (int r = 0; r < 8; ++r)
    u[r] = *(const unsigned int*)(base + k0 + frag_k(r, half));
  return u;
}

static __device__ __forceinline__ v8f wmma_bf16(v8u a, v8u b, v8f c) {
  return __builtin_amdgcn_wmma_f32_16x16x32_bf16(
      false, __builtin_bit_cast(v16bf, a), false, __builtin_bit_cast(v16bf, b),
      (short)0, c, false, false);
}

// -------------------------- elementwise / prep kernels ----------------------

__global__ void k_zero(float* p, int n) {
  int i = blockIdx.x * blockDim.x + threadIdx.x;
  if (i < n) p[i] = 0.f;
}

// xw = x @ W ; layout (B,N,H,C), o = h*C+c
__global__ void k_xw(const float* x, const float* W, float* xw,
                     int B, int N, int ci, int H, int C) {
  int idx = blockIdx.x * blockDim.x + threadIdx.x;
  int tot = B * N * H * C;
  if (idx >= tot) return;
  int HC = H * C;
  int o = idx % HC;
  int n = (idx / HC) % N;
  int b = idx / (HC * N);
  const float* xr = x + ((size_t)b * N + n) * ci;
  float s = 0.f;
  for (int k = 0; k < ci; ++k) s += xr[k] * W[k * HC + o];
  xw[idx] = s;
}

// xwT bf16 (B*H, C, Np): K-contiguous over source node j, zero-padded
__global__ void k_xwT(const float* xw, __bf16* xwT,
                      int B, int N, int H, int C, int Np) {
  int idx = blockIdx.x * blockDim.x + threadIdx.x;
  int tot = B * H * C * Np;
  if (idx >= tot) return;
  int n = idx % Np;
  int c = (idx / Np) % C;
  int h = (idx / (Np * C)) % H;
  int b = idx / (Np * C * H);
  float v = (n < N) ? xw[(((size_t)b * N + n) * H + h) * C + c] : 0.f;
  xwT[idx] = (__bf16)v;
}

// a_src[b,n,h] = dot(xw[b,n,h,:], asrcW[h,:]) ; same for a_dst
__global__ void k_coef(const float* xw, const float* asrcW, const float* adstW,
                       float* asrc, float* adst, int B, int N, int H, int C) {
  int idx = blockIdx.x * blockDim.x + threadIdx.x;
  int tot = B * N * H;
  if (idx >= tot) return;
  int h = idx % H;
  const float* p = xw + (size_t)idx * C;
  float s = 0.f, d = 0.f;
  for (int c = 0; c < C; ++c) {
    s += p[c] * asrcW[h * C + c];
    d += p[c] * adstW[h * C + c];
  }
  asrc[idx] = s;
  adst[idx] = d;
}

// Masked-softmax row stats (max, sum) over source nodes. One wave per row.
__global__ void k_stats(const float* adj, const float* asrc, const float* adst,
                        float* rowmax, float* rowsum, int B, int N, int H) {
  int wid = threadIdx.x >> 5, lane = threadIdx.x & 31;
  int row = blockIdx.x * (blockDim.x >> 5) + wid;
  int tot = B * H * N;
  if (row >= tot) return;
  int i = row % N;
  int h = (row / N) % H;
  int b = row / (N * H);
  const float* arow = adj + ((size_t)b * N + i) * N;
  float ad = adst[((size_t)b * N + i) * H + h];
  float mx = -1e30f;
  for (int j = lane; j < N; j += 32) {
    float av = arow[j];
    float e = asrc[((size_t)b * N + j) * H + h] + ad;
    e = e > 0.f ? e : 0.2f * e;
    bool conn = (av != 0.f) || (j == i);
    mx = fmaxf(mx, conn ? e : -1e30f);
  }
  mx = waveMax(mx);
  float sm = 0.f;
  for (int j = lane; j < N; j += 32) {
    float av = arow[j];
    float e = asrc[((size_t)b * N + j) * H + h] + ad;
    e = e > 0.f ? e : 0.2f * e;
    bool conn = (av != 0.f) || (j == i);
    sm += conn ? __expf(e - mx) : 0.f;
  }
  sm = waveSum(sm);
  if (lane == 0) {
    rowmax[row] = mx;
    rowsum[row] = sm;
  }
}

// Materialize normalized attention probabilities P bf16 (B*H, N, Np),
// mask folded in, branchless, two j per thread (one dword store).
__global__ void k_prob(const float* adj, const float* asrc, const float* adst,
                       const float* rowmax, const float* rowsum,
                       __bf16* P, int B, int N, int H, int Np) {
  int hp = Np >> 1;
  int idx = blockIdx.x * blockDim.x + threadIdx.x;
  int tot = B * H * N * hp;
  if (idx >= tot) return;
  int jp = (idx % hp) * 2;
  int i = (idx / hp) % N;
  int h = (idx / hp / N) % H;
  int b = idx / hp / N / H;
  int rowid = ((b * H + h) * N) + i;
  float rmx = rowmax[rowid];
  float rs = rowsum[rowid];
  float inv = rs > 0.f ? 1.f / rs : 0.f;
  float ad = adst[((size_t)b * N + i) * H + h];
  unsigned int outw = 0;
#pragma unroll
  for (int p = 0; p < 2; ++p) {
    int j = jp + p;
    float pv = 0.f;
    if (j < N) {
      float av = adj[((size_t)b * N + i) * N + j];
      float e = asrc[((size_t)b * N + j) * H + h] + ad;
      e = e > 0.f ? e : 0.2f * e;
      float ev = __expf(e - rmx) * inv;
      pv = ((av != 0.f) || (j == i)) ? ev : 0.f;
    }
    __bf16 hb = (__bf16)pv;
    outw |= ((unsigned int)__builtin_bit_cast(unsigned short, hb)) << (16 * p);
  }
  *(unsigned int*)(P + ((((size_t)b * H + h) * N + i) * Np + jp)) = outw;
}

// fp32 (BATCH,R,K) -> bf16 (BATCH,R,Kp) zero-padded in K
__global__ void k_cvt(const float* in, __bf16* out, int BATCH, int R, int K, int Kp) {
  int idx = blockIdx.x * blockDim.x + threadIdx.x;
  int tot = BATCH * R * Kp;
  if (idx >= tot) return;
  int k = idx % Kp;
  int r = (idx / Kp) % R;
  int bb = idx / (Kp * R);
  float v = (k < K) ? in[((size_t)bb * R + r) * K + k] : 0.f;
  out[idx] = (__bf16)v;
}

// fp32 (BATCH,R,C) -> bf16 (BATCH,C,Rp) transposed, zero-padded in Rp
__global__ void k_cvtT(const float* in, __bf16* out, int BATCH, int R, int C, int Rp) {
  int idx = blockIdx.x * blockDim.x + threadIdx.x;
  int tot = BATCH * C * Rp;
  if (idx >= tot) return;
  int rp = idx % Rp;
  int c = (idx / Rp) % C;
  int bb = idx / (Rp * C);
  float v = (rp < R) ? in[((size_t)bb * R + rp) * C + c] : 0.f;
  out[idx] = (__bf16)v;
}

// ------------------------------ WMMA kernels --------------------------------

// C[b](MxNc) f32 = A[b](MxKp) bf16 row-major x B[b](NcxKp) bf16 row-major^T.
// 32x32 output block per wave (2x2 WMMA tiles), software-pipelined K loop.
// Kp must be a multiple of 32 (operands zero-padded). 4 waves / block.
__global__ void k_gemm_tn(const __bf16* A, const __bf16* Bm, float* Cm,
                          int BATCH, int M, int Kp, int Nc,
                          long long sA, long long sB, long long sC) {
  int mt = (M + 31) / 32, nt = (Nc + 31) / 32;
  int tiles = BATCH * mt * nt;
  int wid = threadIdx.x >> 5;
  int tile = blockIdx.x * (blockDim.x >> 5) + wid;
  if (tile >= tiles) return;                 // wave-uniform
  int ntb = tile % nt; tile /= nt;
  int mtb = tile % mt;
  int bb = tile / mt;
  int lane = threadIdx.x & 31, half = lane >> 4, lo = lane & 15;
  int m0 = mtb * 32 + lo, m1 = m0 + 16;
  int n0 = ntb * 32 + lo, n1 = n0 + 16;
  int m0c = m0 < M ? m0 : M - 1;             // clamp: garbage rows never stored
  int m1c = m1 < M ? m1 : M - 1;
  int n0c = n0 < Nc ? n0 : Nc - 1;
  int n1c = n1 < Nc ? n1 : Nc - 1;
  const __bf16* a0p = A + (size_t)bb * sA + (size_t)m0c * Kp;
  const __bf16* a1p = A + (size_t)bb * sA + (size_t)m1c * Kp;
  const __bf16* b0p = Bm + (size_t)bb * sB + (size_t)n0c * Kp;
  const __bf16* b1p = Bm + (size_t)bb * sB + (size_t)n1c * Kp;
  v8f acc00 = {}, acc01 = {}, acc10 = {}, acc11 = {};
  // prologue: chunk 0 in flight
  v8u a0 = load_frag(a0p, 0, half);
  v8u a1 = load_frag(a1p, 0, half);
  v8u b0 = load_frag(b0p, 0, half);
  v8u b1 = load_frag(b1p, 0, half);
  for (int k0 = 0; k0 < Kp; k0 += 32) {
    // issue next chunk's loads before consuming current chunk (pipelined);
    // last iteration harmlessly re-loads chunk 0 to stay branchless.
    int kn = (k0 + 32 < Kp) ? (k0 + 32) : 0;
    v8u na0 = load_frag(a0p, kn, half);
    v8u na1 = load_frag(a1p, kn, half);
    v8u nb0 = load_frag(b0p, kn, half);
    v8u nb1 = load_frag(b1p, kn, half);
    acc00 = wmma_bf16(a0, b0, acc00);
    acc01 = wmma_bf16(a0, b1, acc01);
    acc10 = wmma_bf16(a1, b0, acc10);
    acc11 = wmma_bf16(a1, b1, acc11);
    a0 = na0; a1 = na1; b0 = nb0; b1 = nb1;
  }
  float* cb = Cm + (size_t)bb * sC;
#pragma unroll
  for (int v = 0; v < 8; ++v) {
    int r0 = mtb * 32 + v + 8 * half;        // C/D layout: M = v + 8*half
    int r1 = r0 + 16;
    if (r0 < M && n0 < Nc) cb[(size_t)r0 * Nc + n0] = acc00[v];
    if (r0 < M && n1 < Nc) cb[(size_t)r0 * Nc + n1] = acc01[v];
    if (r1 < M && n0 < Nc) cb[(size_t)r1 * Nc + n0] = acc10[v];
    if (r1 < M && n1 < Nc) cb[(size_t)r1 * Nc + n1] = acc11[v];
  }
}

// sum over (b,i,j) of (adj - s s^T)^2, s bf16 (B,N,Kp) K-contiguous.
__global__ void k_link(const float* adj, const __bf16* s, float* acc,
                       int B, int N, int Kp) {
  int it = (N + 15) / 16;
  int tiles = B * it * it;
  int wid = threadIdx.x >> 5;
  int tile = blockIdx.x * (blockDim.x >> 5) + wid;
  if (tile >= tiles) return;                 // wave-uniform
  int jt = tile % it; tile /= it;
  int mt = tile % it;
  int b = tile / it;
  int lane = threadIdx.x & 31, half = lane >> 4, lo = lane & 15;
  int i0 = mt * 16, j0 = jt * 16;
  int ic = i0 + lo < N ? i0 + lo : N - 1;
  int jc = j0 + lo < N ? j0 + lo : N - 1;
  const __bf16* ar = s + ((size_t)b * N + ic) * Kp;
  const __bf16* br = s + ((size_t)b * N + jc) * Kp;
  v8f accf = {};
  v8u ua = load_frag(ar, 0, half);
  v8u ub = load_frag(br, 0, half);
  for (int k0 = 0; k0 < Kp; k0 += 32) {
    int kn = (k0 + 32 < Kp) ? (k0 + 32) : 0;
    v8u na = load_frag(ar, kn, half);
    v8u nb = load_frag(br, kn, half);
    accf = wmma_bf16(ua, ub, accf);
    ua = na; ub = nb;
  }
  float ss = 0.f;
#pragma unroll
  for (int v = 0; v < 8; ++v) {
    int i = i0 + v + 8 * half, j = j0 + lo;
    if (i < N && j < N) {
      float d = adj[((size_t)b * N + i) * N + j] - accf[v];
      ss += d * d;
    }
  }
  ss = waveSum(ss);
  if (lane == 0) atomicAdd(acc, ss);
}

// ------------------------------ misc kernels --------------------------------

__global__ void k_headmean(const float* aggr, const float* bias, float* y,
                           int B, int N, int H, int C) {
  int idx = blockIdx.x * blockDim.x + threadIdx.x;
  int tot = B * N * C;
  if (idx >= tot) return;
  int c = idx % C;
  int n = (idx / C) % N;
  int b = idx / (C * N);
  float s = 0.f;
  for (int h = 0; h < H; ++h) s += aggr[(((size_t)b * H + h) * N + n) * C + c];
  y[idx] = s / (float)H + bias[c];
}

__global__ void k_graphnorm(float* x, const float* w, const float* bias,
                            const float* ms, int B, int N, int C) {
  int b = blockIdx.x / C, c = blockIdx.x % C;
  __shared__ float sh[256];
  int t = threadIdx.x;
  float s = 0.f;
  for (int n = t; n < N; n += 256) s += x[((size_t)b * N + n) * C + c];
  sh[t] = s; __syncthreads();
  for (int o = 128; o > 0; o >>= 1) { if (t < o) sh[t] += sh[t + o]; __syncthreads(); }
  float mean = sh[0] / (float)N; __syncthreads();
  float msc = ms[c];
  float v = 0.f;
  for (int n = t; n < N; n += 256) {
    float o = x[((size_t)b * N + n) * C + c] - msc * mean;
    v += o * o;
  }
  sh[t] = v; __syncthreads();
  for (int o = 128; o > 0; o >>= 1) { if (t < o) sh[t] += sh[t + o]; __syncthreads(); }
  float var = sh[0] / (float)N;
  float sc = w[c] * rsqrtf(var + 1e-5f);
  float bb = bias[c];
  for (int n = t; n < N; n += 256) {
    size_t id = ((size_t)b * N + n) * C + c;
    x[id] = (x[id] - msc * mean) * sc + bb;
  }
}

__global__ void k_softmax_row(const float* x, float* y, int rows, int K) {
  int wid = threadIdx.x >> 5, lane = threadIdx.x & 31;
  int row = blockIdx.x * (blockDim.x >> 5) + wid;
  if (row >= rows) return;
  const float* p = x + (size_t)row * K;
  float* q = y + (size_t)row * K;
  float mx = -1e30f;
  for (int k = lane; k < K; k += 32) mx = fmaxf(mx, p[k]);
  mx = waveMax(mx);
  float sm = 0.f;
  for (int k = lane; k < K; k += 32) sm += __expf(p[k] - mx);
  sm = waveSum(sm);
  float inv = 1.f / sm;
  for (int k = lane; k < K; k += 32) q[k] = __expf(p[k] - mx) * inv;
}

__global__ void k_ent(const float* s, float* acc, int rows, int K) {
  int wid = threadIdx.x >> 5, lane = threadIdx.x & 31;
  int row = blockIdx.x * (blockDim.x >> 5) + wid;
  if (row >= rows) return;
  const float* p = s + (size_t)row * K;
  float t = 0.f;
  for (int k = lane; k < K; k += 32) {
    float v = p[k];
    t += -v * logf(v + 1e-15f);
  }
  t = waveSum(t);
  if (lane == 0) atomicAdd(acc, t);
}

__global__ void k_transpose(const float* in, float* out, int B, int T, int K) {
  int idx = blockIdx.x * blockDim.x + threadIdx.x;
  int tot = B * T * K;
  if (idx >= tot) return;
  int k = idx % K;
  int t = (idx / K) % T;
  int b = idx / (K * T);
  out[((size_t)b * K + k) * T + t] = in[idx];
}

__global__ void k_meanrows(const float* in, float* out, int B, int R, int F) {
  int idx = blockIdx.x * blockDim.x + threadIdx.x;
  int tot = B * F;
  if (idx >= tot) return;
  int f = idx % F, b = idx / F;
  float s = 0.f;
  for (int r = 0; r < R; ++r) s += in[((size_t)b * R + r) * F + f];
  out[idx] = s / (float)R;
}

__global__ void k_meanlast(const float* in, float* out, int rows, int T) {
  int i = blockIdx.x * blockDim.x + threadIdx.x;
  if (i >= rows) return;
  float s = 0.f;
  for (int t = 0; t < T; ++t) s += in[(size_t)i * T + t];
  out[i] = s / (float)T;
}

__global__ void k_concat(const float* zx, const float* sx, float* qs,
                         int B, int N1, int F) {
  int idx = blockIdx.x * blockDim.x + threadIdx.x;
  int tot = B * (N1 + 1) * F;
  if (idx >= tot) return;
  int f = idx % F;
  int r = (idx / F) % (N1 + 1);
  int b = idx / (F * (N1 + 1));
  qs[idx] = (r < N1) ? zx[((size_t)b * N1 + r) * F + f] : sx[b * F + f];
}

__global__ void k_finalize(const float* sc, float* o_l, float* o_e,
                           int B, int N, int N1) {
  *o_l = sqrtf(sc[0]) / ((float)B * N * N) + sqrtf(sc[1]) / ((float)B * N1 * N1);
  *o_e = sc[2] / ((float)B * N) + sc[3] / ((float)B * N1);
}

// ------------------------------ host driver ---------------------------------

static inline int cdiv(int a, int b) { return (a + b - 1) / b; }

struct ConvP { const float *W, *adst, *asrc, *bias; int C; };
struct NormP { const float *b, *ms, *w; };
struct GnnP  { ConvP c1, c2; NormP n1, n2; };

static void gemm(hipStream_t st, const __bf16* A, const __bf16* Bm, float* Cm,
                 int BATCH, int M, int Kp, int Nc,
                 long long sA, long long sB, long long sC) {
  int mt = cdiv(M, 32), nt = cdiv(Nc, 32);
  int tiles = BATCH * mt * nt;
  k_gemm_tn<<<cdiv(tiles, 4), 128, 0, st>>>(A, Bm, Cm, BATCH, M, Kp, Nc, sA, sB, sC);
}

static void run_conv(hipStream_t st, const float* x, const float* adj,
                     int B, int N, int Np, int ci, const ConvP& cp,
                     float* xw, __bf16* xwT, float* asr, float* adt,
                     float* rmx, float* rsm, __bf16* Pbf, float* aggr, float* y) {
  const int H = 2, C = cp.C;
  k_xw<<<cdiv(B * N * H * C, 256), 256, 0, st>>>(x, cp.W, xw, B, N, ci, H, C);
  k_xwT<<<cdiv(B * H * C * Np, 256), 256, 0, st>>>(xw, xwT, B, N, H, C, Np);
  k_coef<<<cdiv(B * N * H, 128), 128, 0, st>>>(xw, cp.asrc, cp.adst, asr, adt, B, N, H, C);
  k_stats<<<cdiv(B * H * N, 4), 128, 0, st>>>(adj, asr, adt, rmx, rsm, B, N, H);
  k_prob<<<cdiv(B * H * N * (Np / 2), 256), 256, 0, st>>>(adj, asr, adt, rmx, rsm,
                                                          Pbf, B, N, H, Np);
  gemm(st, Pbf, xwT, aggr, B * H, N, Np, C,
       (long long)N * Np, (long long)C * Np, (long long)N * C);
  k_headmean<<<cdiv(B * N * C, 256), 256, 0, st>>>(aggr, cp.bias, y, B, N, H, C);
}

static void run_norm(hipStream_t st, float* x, const NormP& np, int B, int N, int C) {
  k_graphnorm<<<B * C, 256, 0, st>>>(x, np.w, np.b, np.ms, B, N, C);
}

static void run_gnn(hipStream_t st, const float* x, const float* adj,
                    int B, int N, int Np, int ci, const GnnP& g,
                    float* xw, __bf16* xwT, float* asr, float* adt,
                    float* rmx, float* rsm, __bf16* Pbf, float* aggr,
                    float* hid, float* out) {
  run_conv(st, x, adj, B, N, Np, ci, g.c1, xw, xwT, asr, adt, rmx, rsm, Pbf, aggr, hid);
  run_norm(st, hid, g.n1, B, N, g.c1.C);
  run_conv(st, hid, adj, B, N, Np, g.c1.C, g.c2, xw, xwT, asr, adt, rmx, rsm, Pbf, aggr, out);
  run_norm(st, out, g.n2, B, N, g.c2.C);
}

extern "C" void kernel_launch(void* const* d_in, const int* in_sizes, int n_in,
                              void* d_out, int out_size, void* d_ws, size_t ws_size,
                              hipStream_t stream) {
  const int B = 8, N = 1024, F = 64, T = 64, N1 = 100, N2 = 10, H = 2;
  const int N1p = 128, N2p = 32, Np2 = 128;  // padded K dims (multiples of 32)
  (void)in_sizes; (void)n_in; (void)out_size; (void)ws_size; (void)H;

  const float* x    = (const float*)d_in[0];
  const float* adj  = (const float*)d_in[1];
  const float* oseq = (const float*)d_in[2];

  // params: jax tree-flatten order (sorted dict keys):
  // gnn1_embed, gnn1_pool, gnn2_embed, gnn2_pool; within each:
  // conv1{W,a_dst,a_src,b}, conv2{...}, norm1{b,ms,w}, norm2{b,ms,w}
  int pi = 3;
  auto nextp = [&]() -> const float* { return (const float*)d_in[pi++]; };
  auto readGnn = [&](int c2C) -> GnnP {
    GnnP g;
    g.c1.W = nextp(); g.c1.adst = nextp(); g.c1.asrc = nextp(); g.c1.bias = nextp(); g.c1.C = 64;
    g.c2.W = nextp(); g.c2.adst = nextp(); g.c2.asrc = nextp(); g.c2.bias = nextp(); g.c2.C = c2C;
    g.n1.b = nextp(); g.n1.ms = nextp(); g.n1.w = nextp();
    g.n2.b = nextp(); g.n2.ms = nextp(); g.n2.w = nextp();
    return g;
  };
  GnnP g1e = readGnn(F);
  GnnP g1p = readGnn(N1);
  GnnP g2e = readGnn(F);
  GnnP g2p = readGnn(N2);

  // workspace arena (256B-aligned chunks)
  char* base = (char*)d_ws;
  size_t off = 0;
  auto alloc = [&](size_t bytes) -> void* {
    void* p = base + off;
    off = (off + bytes + 255) & ~(size_t)255;
    return p;
  };
  float*  xw      = (float*)alloc((size_t)B * N * 2 * 100 * 4);
  float*  aggr    = (float*)alloc((size_t)B * N * 2 * 100 * 4);
  float*  asr     = (float*)alloc((size_t)B * N * 2 * 4);
  float*  adt     = (float*)alloc((size_t)B * N * 2 * 4);
  float*  rmx     = (float*)alloc((size_t)B * 2 * N * 4);
  float*  rsm     = (float*)alloc((size_t)B * 2 * N * 4);
  float*  hid     = (float*)alloc((size_t)B * N * 64 * 4);
  float*  n2z_s   = (float*)alloc((size_t)B * N * N1 * 4);
  float*  n2z_x   = (float*)alloc((size_t)B * N * F * 4);
  float*  s1      = (float*)alloc((size_t)B * N * N1 * 4);
  float*  tbuf    = (float*)alloc((size_t)B * N * N1 * 4);
  float*  adj_z   = (float*)alloc((size_t)B * N1 * N1 * 4);
  float*  zt      = (float*)alloc((size_t)B * T * N1 * 4);
  float*  z2s_s   = (float*)alloc((size_t)B * N1 * N2 * 4);
  float*  z2s_x   = (float*)alloc((size_t)B * N1 * F * 4);
  float*  s2      = (float*)alloc((size_t)B * N1 * N2 * 4);
  float*  pooled2 = (float*)alloc((size_t)B * N2 * F * 4);
  float*  stpre   = (float*)alloc((size_t)B * N2 * T * 4);
  float*  scal    = (float*)alloc(32);         // [link1, link2, ent1, ent2]
  __bf16* Pbf     = (__bf16*)alloc((size_t)B * 2 * N * N * 2);
  __bf16* xwT     = (__bf16*)alloc((size_t)B * 2 * 100 * N * 2);
  __bf16* adjbf   = (__bf16*)alloc((size_t)B * N * N * 2);
  __bf16* s1bf    = (__bf16*)alloc((size_t)B * N * N1p * 2);
  __bf16* s1T     = (__bf16*)alloc((size_t)B * N1 * N * 2);
  __bf16* n2z_xT  = (__bf16*)alloc((size_t)B * F * N * 2);
  __bf16* tT      = (__bf16*)alloc((size_t)B * N1 * N * 2);
  __bf16* oseqbf  = (__bf16*)alloc((size_t)B * T * N * 2);
  __bf16* ztbf    = (__bf16*)alloc((size_t)B * T * N1p * 2);
  __bf16* s2bf    = (__bf16*)alloc((size_t)B * N1 * N2p * 2);
  __bf16* s2T     = (__bf16*)alloc((size_t)B * N2 * N1p * 2);
  __bf16* z2s_xT  = (__bf16*)alloc((size_t)B * F * N1p * 2);

  // output regions (flat, in tuple return order)
  float* out      = (float*)d_out;
  float* o_l      = out;
  float* o_e      = out + 1;
  float* o_zone_x = out + 2;
  float* o_zt     = o_zone_x + (size_t)B * N1 * F;
  float* o_sx     = o_zt + (size_t)B * N1 * T;
  float* o_st     = o_sx + (size_t)B * F;
  float* o_qs     = o_st + (size_t)B * N2;

  k_zero<<<1, 8, 0, stream>>>(scal, 8);

  // ---- level 1 GNNs on (B,N) graph (Np = N = 1024, already mult of 32) ----
  run_gnn(stream, x, adj, B, N, N, F, g1p, xw, xwT, asr, adt, rmx, rsm, Pbf, aggr, hid, n2z_s);
  run_gnn(stream, x, adj, B, N, N, F, g1e, xw, xwT, asr, adt, rmx, rsm, Pbf, aggr, hid, n2z_x);

  // ---- diff-pool level 1 ----
  k_softmax_row<<<cdiv(B * N, 4), 128, 0, stream>>>(n2z_s, s1, B * N, N1);
  k_cvt<<<cdiv(B * N * N1p, 256), 256, 0, stream>>>(s1, s1bf, B, N, N1, N1p);
  k_cvtT<<<cdiv(B * N1 * N, 256), 256, 0, stream>>>(s1, s1T, B, N, N1, N);
  k_cvtT<<<cdiv(B * F * N, 256), 256, 0, stream>>>(n2z_x, n2z_xT, B, N, F, N);
  k_cvt<<<cdiv(B * N * N, 256), 256, 0, stream>>>(adj, adjbf, B, N, N, N);
  k_cvt<<<cdiv(B * T * N, 256), 256, 0, stream>>>(oseq, oseqbf, B, T, N, N);
  // zone_x = s^T x
  gemm(stream, s1T, n2z_xT, o_zone_x, B, N1, N, F,
       (long long)N1 * N, (long long)F * N, (long long)N1 * F);
  // t = adj s
  gemm(stream, adjbf, s1T, tbuf, B, N, N, N1,
       (long long)N * N, (long long)N1 * N, (long long)N * N1);
  k_cvtT<<<cdiv(B * N1 * N, 256), 256, 0, stream>>>(tbuf, tT, B, N, N1, N);
  // adj_z = s^T t
  gemm(stream, s1T, tT, adj_z, B, N1, N, N1,
       (long long)N1 * N, (long long)N1 * N, (long long)N1 * N1);
  {
    int it = cdiv(N, 16);
    k_link<<<cdiv(B * it * it, 4), 128, 0, stream>>>(adj, s1bf, scal + 0, B, N, N1p);
  }
  k_ent<<<cdiv(B * N, 4), 128, 0, stream>>>(s1, scal + 2, B * N, N1);
  // zone_temp (B,T,N1)
  gemm(stream, oseqbf, s1T, zt, B, T, N, N1,
       (long long)T * N, (long long)N1 * N, (long long)T * N1);

  // ---- level 2 GNNs on pooled (B,N1) graph (Np2 = 128) ----
  run_gnn(stream, o_zone_x, adj_z, B, N1, Np2, F, g2p, xw, xwT, asr, adt, rmx, rsm, Pbf, aggr, hid, z2s_s);
  run_gnn(stream, o_zone_x, adj_z, B, N1, Np2, F, g2e, xw, xwT, asr, adt, rmx, rsm, Pbf, aggr, hid, z2s_x);

  // ---- diff-pool level 2 ----
  k_softmax_row<<<cdiv(B * N1, 4), 128, 0, stream>>>(z2s_s, s2, B * N1, N2);
  k_cvt<<<cdiv(B * N1 * N2p, 256), 256, 0, stream>>>(s2, s2bf, B, N1, N2, N2p);
  k_cvtT<<<cdiv(B * N2 * N1p, 256), 256, 0, stream>>>(s2, s2T, B, N1, N2, N1p);
  k_cvtT<<<cdiv(B * F * N1p, 256), 256, 0, stream>>>(z2s_x, z2s_xT, B, N1, F, N1p);
  gemm(stream, s2T, z2s_xT, pooled2, B, N2, N1p, F,
       (long long)N2 * N1p, (long long)F * N1p, (long long)N2 * F);
  {
    int it = cdiv(N1, 16);
    k_link<<<cdiv(B * it * it, 4), 128, 0, stream>>>(adj_z, s2bf, scal + 1, B, N1, N2p);
  }
  k_ent<<<cdiv(B * N1, 4), 128, 0, stream>>>(s2, scal + 3, B * N1, N2);

  // semantic_x = mean_rows(pooled2) (B,1,F)
  k_meanrows<<<cdiv(B * F, 256), 256, 0, stream>>>(pooled2, o_sx, B, N2, F);
  // semantic_temp pre: (B,N2,T) = s2^T x zone_temp^T ; then mean over T
  k_cvt<<<cdiv(B * T * N1p, 256), 256, 0, stream>>>(zt, ztbf, B, T, N1, N1p);
  gemm(stream, s2T, ztbf, stpre, B, N2, N1p, T,
       (long long)N2 * N1p, (long long)T * N1p, (long long)N2 * T);
  k_meanlast<<<cdiv(B * N2, 64), 64, 0, stream>>>(stpre, o_st, B * N2, T);
  // zone_temp output = swapaxes(zt,1,2) -> (B,N1,T)
  k_transpose<<<cdiv(B * T * N1, 256), 256, 0, stream>>>(zt, o_zt, B, T, N1);
  // qs = concat(zone_x, semantic_x)
  k_concat<<<cdiv(B * (N1 + 1) * F, 256), 256, 0, stream>>>(o_zone_x, o_sx, o_qs, B, N1, F);
  // losses
  k_finalize<<<1, 1, 0, stream>>>(scal, o_l, o_e, B, N, N1);
}
